// HadamardTransform_71356586655955
// MI455X (gfx1250) — matrix operand order repaired and checked by reading, not compiled, not run on hardware
//
#include <hip/hip_runtime.h>

typedef _Float16 v16h __attribute__((ext_vector_type(16)));
typedef _Float16 v8h  __attribute__((ext_vector_type(8)));
typedef float    v8f  __attribute__((ext_vector_type(8)));
typedef unsigned v4u  __attribute__((ext_vector_type(4)));

#define DIMC 768
#define SP   3136            // 56*56 spatial positions per (b, c)
#define NS   32              // spatial columns per workgroup
#define C1N  24              // 768 / 32
#define IN_CS 24             // halves per In chunk: 16 data + 8 pad (48B, 16B-aligned)
#define T_CS  40             // halves per T chunk: 32 data + 8 pad (80B, 16B-aligned)
#define IN_LDS_BYTES (48 * 32 * IN_CS * 2)   // 73728
#define T_LDS_BYTES  (1024 * T_CS * 2)       // 81920
#define LDS_BYTES    (IN_LDS_BYTES + T_LDS_BYTES)  // 155648 -> 2 WGs / WGP

__global__ __launch_bounds__(256)
void hadamard1024_kron_kernel(const float* __restrict__ x, float* __restrict__ out)
{
    extern __shared__ char smem[];
    _Float16* inlds = (_Float16*)smem;                   // chunk(cc,s): 16 halves over c%16
    _Float16* tlds  = (_Float16*)(smem + IN_LDS_BYTES);  // chunk(j0,s): 32 halves over c1

    const int tid  = threadIdx.x;
    const int lane = tid & 31;
    const int wave = tid >> 5;
    const int n_   = lane & 15;   // tile column / A-row
    const int g    = lane >> 4;   // lane group (K-half select)

    const int tile  = blockIdx.x;
    const int b     = blockIdx.y;
    const int sbase = tile * NS;

    // ---- Hadamard A-fragments in registers (exact in f16) ----
    // A 16x32 f16 fragment, contiguous k order used here: lane row m = lane&15; half i -> K = i + 16g
    v16h a1[2], a2[2];
    {
        const int m = n_;
#pragma unroll
        for (int t = 0; t < 2; ++t) {
#pragma unroll
            for (int i = 0; i < 16; ++i) {
                const int k  = i + 16 * g;
                const int jr = 16 * t + m;
                const int s1 = __popc((unsigned)(jr & k)) & 1;
                a1[t][i] = s1 ? (_Float16)(-1.0f) : (_Float16)(1.0f);
                const float av2 = (k < C1N) ? (s1 ? -0.03125f : 0.03125f) : 0.0f;
                a2[t][i] = (_Float16)av2;   // carries 1/32 scale; K-pad (c1>=24) = 0
            }
        }
    }

    // ---- Phase A: zero T K-pad (c1=24..31 of every chunk), stream In -> swizzled LDS ----
    {
        const v4u z = {0u, 0u, 0u, 0u};
        for (int chunk = tid; chunk < 1024; chunk += 256)
            *(v4u*)(tlds + chunk * T_CS + 24) = z;       // halves 24..31
    }
    {
        const int s  = tid & 31;
        const int rw = tid >> 5;                          // 0..7
        const float* src = x + (size_t)b * DIMC * SP + sbase + s;
#pragma unroll
        for (int k = 0; k < 6; ++k) {
            const int cc = rw + k * 8;                    // channel-chunk 0..47
            const float* p = src + (size_t)cc * 16 * SP;
            v16h vals;
#pragma unroll
            for (int i = 0; i < 16; ++i)
                vals[i] = (_Float16)__builtin_nontemporal_load(p + (size_t)i * SP);
            _Float16* q = inlds + (cc * 32 + s) * IN_CS;
            *(v8h*)q       = __builtin_shufflevector(vals, vals, 0,1,2,3,4,5,6,7);
            *(v8h*)(q + 8) = __builtin_shufflevector(vals, vals, 8,9,10,11,12,13,14,15);
        }
    }
    __syncthreads();

    // ---- Stage 1: T[(j0,s)][c1] = sum_{c0} A1[j0,c0] * In[32c1+c0][s] ----
    for (int c1 = wave; c1 < C1N; c1 += 8) {
        v16h bf[2];
#pragma unroll
        for (int nt = 0; nt < 2; ++nt) {
            // B half i -> K = c0 = i + 16g  ==> chunk (2c1+g, nt*16+n_), halves 0..15
            const _Float16* q = inlds + ((c1 * 2 + g) * 32 + nt * 16 + n_) * IN_CS;
            v8h lo = *(const v8h*)q;
            v8h hi = *(const v8h*)(q + 8);
            bf[nt] = __builtin_shufflevector(lo, hi, 0,1,2,3,4,5,6,7,8,9,10,11,12,13,14,15);
        }
#pragma unroll
        for (int t = 0; t < 2; ++t) {
#pragma unroll
            for (int nt = 0; nt < 2; ++nt) {
                v8f acc = {};
                acc = __builtin_amdgcn_wmma_f32_16x16x32_f16(
                        false, a1[t], false, bf[nt], (short)0, acc, false, false);
#pragma unroll
                for (int v = 0; v < 8; ++v) {
                    const int j0 = 16 * t + v + 8 * g;    // D: VGPR v -> M = v + 8g
                    tlds[(j0 * 32 + nt * 16 + n_) * T_CS + c1] = (_Float16)acc[v];
                }
            }
        }
    }
    __syncthreads();

    // ---- Stage 2: Out[32j1+j0][s] = sum_{c1} A2[j1,c1] * T[(j0,s)][c1] ----
    float* dst = out + (size_t)b * DIMC * SP + sbase;
#pragma unroll
    for (int qq = 0; qq < 8; ++qq) {
        const int nt = wave * 8 + qq;     // 64 n-tiles over (j0, s-half)
        const int j0 = nt >> 1;
        const int sh = (nt & 1) * 16;
        // B half i -> K = c1 = i + 16g ==> chunk (j0, sh+n_), halves 16g..16g+15
        const _Float16* p = tlds + (j0 * 32 + sh + n_) * T_CS + 16 * g;
        v8h lo = *(const v8h*)p;
        v8h hi = *(const v8h*)(p + 8);
        v16h bf = __builtin_shufflevector(lo, hi, 0,1,2,3,4,5,6,7,8,9,10,11,12,13,14,15);

        v8f d0 = {}, d1 = {};
        d0 = __builtin_amdgcn_wmma_f32_16x16x32_f16(false, a2[0], false, bf, (short)0, d0, false, false);
        d1 = __builtin_amdgcn_wmma_f32_16x16x32_f16(false, a2[1], false, bf, (short)0, d1, false, false);
#pragma unroll
        for (int v = 0; v < 8; ++v) {
            const int j1 = v + 8 * g;                      // 0..15, all valid
            __builtin_nontemporal_store(d0[v], dst + (size_t)(32 * j1 + j0) * SP + sh + n_);
        }
        if (g == 0) {                                      // j1 = 16+v (16..23 valid only)
#pragma unroll
            for (int v = 0; v < 8; ++v) {
                const int j1 = 16 + v;
                __builtin_nontemporal_store(d1[v], dst + (size_t)(32 * j1 + j0) * SP + sh + n_);
            }
        }
    }
}

extern "C" void kernel_launch(void* const* d_in, const int* in_sizes, int n_in,
                              void* d_out, int out_size, void* d_ws, size_t ws_size,
                              hipStream_t stream) {
    const float* x = (const float*)d_in[0];
    float* out = (float*)d_out;
    const int B = in_sizes[0] / (DIMC * SP);   // 16
    dim3 grid(SP / NS, B);                     // 98 x 16 workgroups
    hadamard1024_kron_kernel<<<grid, dim3(256), LDS_BYTES, stream>>>(x, out);
}
